// PixelRNN_7765300871325
// MI455X (gfx1250) — matrix operand order repaired and checked by reading, not compiled, not run on hardware
//
#include <hip/hip_runtime.h>

#define NLAY 3
#define BT   16
#define WW   28
#define HH   28
#define TPB  128

typedef __attribute__((ext_vector_type(16))) __bf16 v16bf;
typedef __attribute__((ext_vector_type(8)))  float  v8f;

union Frag { v16bf v; unsigned u[8]; };

// All on-chip state for one 16-image tile: 253,896 B of the 320KB WGP LDS.
struct Smem {
  unsigned short Wl[NLAY][64][96];          // gate weights bf16: k<48 is_w (tap2 masked), k>=48 ss_w
  unsigned short CWB[16][32];               // masked 7x7 conv as B-matrix [o][k = kh*8+kw], kw7/kh>3 zero
  unsigned short H[NLAY][2][BT][WW+2][16];  // hidden state bf16, row double-buffered, w zero-padded
  unsigned short IN[2][BT][WW+2][16];       // per-row layer-input ping-pong (bf16, w zero-padded)
  unsigned short XR[4][BT][36];             // rolling x rows r-3..r (bf16, 3/5 col zero pad)
  float          C[NLAY][BT][WW][16];       // LSTM cell state fp32
  float          biasL[NLAY][64];           // is_b + ss_b
  float          convb[16];
  float          c1w[16][16];
  float          c1b[16];
  float          c2w[16];
  float          c2b;
  float          pad0;
  float          red[TPB];
};

__device__ __forceinline__ unsigned short f2bf(float f) {
  unsigned u = __float_as_uint(f);
  u += 0x7fffu + ((u >> 16) & 1u);          // round-to-nearest-even
  return (unsigned short)(u >> 16);
}
__device__ __forceinline__ float bf2f(unsigned short h) {
  return __uint_as_float(((unsigned)h) << 16);
}
__device__ __forceinline__ float sigm(float x)  { return 1.0f / (1.0f + __expf(-x)); }
__device__ __forceinline__ float tanhx(float x) { return 2.0f / (1.0f + __expf(-2.0f * x)) - 1.0f; }

// B fragment (same K order as A: V0-3 K0-7|8-15, V4-7 K16-23|24-31, 2 bf16/VGPR)
__device__ __forceinline__ void load_bfrag(Frag& f, const unsigned short* __restrict__ row,
                                           int ks, int half) {
#pragma unroll
  for (int g = 0; g < 8; ++g) {
    int k = (ks << 5) + ((g >> 2) << 4) + (half << 3) + ((g & 3) << 1);
    f.u[g] = *reinterpret_cast<const unsigned*>(row + k);
  }
}

// A fragment (ISA 16-bit A 16x32: V0-3 K0-7|8-15, V4-7 K16-23|24-31), K=96 split x(0..47)/h(48..95)
__device__ __forceinline__ void load_afrag_lstm(Frag& f,
                                                const unsigned short* __restrict__ xb,
                                                const unsigned short* __restrict__ hb,
                                                int ks, int half) {
#pragma unroll
  for (int g = 0; g < 8; ++g) {
    int k = (ks << 5) + ((g >> 2) << 4) + (half << 3) + ((g & 3) << 1);
    const unsigned short* p = (k < 48) ? (xb + k) : (hb + (k - 48));
    f.u[g] = *reinterpret_cast<const unsigned*>(p);
  }
}

// One w-position of a row-LSTM step: prefetch next-w A frags (double buffer), 12 WMMAs, LSTM update.
__device__ __forceinline__ void lstm_w_step(Smem& S, int l, int rb, int wb, int cur, int nxt,
                                            int chn, int half, int w, int wnext,
                                            const Frag (&bw)[3][4], Frag (&acur)[3], Frag (&anxt)[3])
{
  if (wnext >= 0) {                          // issue next-w fragment loads BEFORE the WMMAs so
    const unsigned short* xb = &S.IN[rb][chn][wnext][0];   // DS latency overlaps the matrix pipe
    const unsigned short* hb = &S.H[l][cur][chn][wnext][0];
#pragma unroll
    for (int ks = 0; ks < 3; ++ks) load_afrag_lstm(anxt[ks], xb, hb, ks, half);
  }
  v8f acc[4];
#pragma unroll
  for (int nt = 0; nt < 4; ++nt) {
    float bv = S.biasL[l][nt * 16 + chn];
#pragma unroll
    for (int i = 0; i < 8; ++i) acc[nt][i] = bv;
  }
#pragma unroll
  for (int ks = 0; ks < 3; ++ks)
#pragma unroll
    for (int nt = 0; nt < 4; ++nt)
      acc[nt] = __builtin_amdgcn_wmma_f32_16x16x32_bf16(false, acur[ks].v, false, bw[ks][nt].v,
                                                        (short)0, acc[nt], false, false);
  // elementwise LSTM update; D layout: element i -> (batch = i+8*half, channel = chn)
#pragma unroll
  for (int i = 0; i < 8; ++i) {
    int b = i + (half << 3);
    float og = sigm(acc[0][i]);
    float fg = sigm(acc[1][i]);
    float ig = sigm(acc[2][i]);
    float gg = tanhx(acc[3][i]);
    float cold = S.C[l][b][w][chn];
    float cnew = fg * cold + ig * gg;
    S.C[l][b][w][chn] = cnew;
    float hv = og * tanhx(cnew);
    S.H[l][nxt][b][w + 1][chn] = f2bf(hv);
    float inv = bf2f(S.IN[rb][b][w + 1][chn]);   // residual: out = h + layer input
    S.IN[wb][b][w + 1][chn] = f2bf(hv + inv);
  }
}

__global__ __launch_bounds__(TPB, 1)
void pixelrnn_fused(const float* __restrict__ xg,
                    const float* __restrict__ conv_w, const float* __restrict__ conv_b,
                    const float* __restrict__ is_w,   const float* __restrict__ is_b,
                    const float* __restrict__ ss_w,   const float* __restrict__ ss_b,
                    const float* __restrict__ h0,     const float* __restrict__ c0,
                    const float* __restrict__ c1_w,   const float* __restrict__ c1_b,
                    const float* __restrict__ c2_w,   const float* __restrict__ c2_b,
                    float* __restrict__ partial)
{
  extern __shared__ char smem_raw[];
  Smem& S = *reinterpret_cast<Smem*>(smem_raw);

  const int tid  = threadIdx.x;
  const int lane = tid & 31;
  const int half = (lane >> 4) & 1;
  const int wv   = tid >> 5;          // wave id 0..3, owns w = wv*7 .. wv*7+6
  const int chn  = lane & 15;         // N-column (channel) / A-row (batch) owned by lane
  const int b0   = blockIdx.x * BT;

  // ---- zero all LDS (pads, H parity-1 buffer, everything) ----
  for (unsigned i = tid; i < (unsigned)(sizeof(Smem) / 4u); i += TPB)
    reinterpret_cast<unsigned*>(&S)[i] = 0u;
  __syncthreads();

  // ---- stage weights into LDS (bf16, K-contiguous per output channel) ----
  for (int idx = tid; idx < NLAY * 64 * 96; idx += TPB) {
    int k = idx % 96, o = (idx / 96) % 64, l = idx / (96 * 64);
    float v;
    if (k < 48) { int t = k / 16, c = k % 16; v = (t == 2) ? 0.0f : is_w[((l * 64 + o) * 16 + c) * 3 + t]; }
    else        { int k2 = k - 48; int t = k2 / 16, c = k2 % 16; v = ss_w[((l * 64 + o) * 16 + c) * 3 + t]; }
    S.Wl[l][o][k] = f2bf(v);
  }
  for (int idx = tid; idx < 16 * 32; idx += TPB) {
    int k = idx % 32, o = idx / 32, kh = k >> 3, kw = k & 7;
    float v = 0.0f;
    if (kw < 7 && (kh < 3 || kw < 4)) v = conv_w[o * 49 + kh * 7 + kw];  // causal mask
    S.CWB[o][k] = f2bf(v);
  }
  for (int idx = tid; idx < NLAY * 64; idx += TPB) {
    int o = idx % 64, l = idx / 64;
    S.biasL[l][o] = is_b[l * 64 + o] + ss_b[l * 64 + o];
  }
  if (tid < 16) { S.convb[tid] = conv_b[tid]; S.c1b[tid] = c1_b[tid]; S.c2w[tid] = c2_w[tid]; }
  for (int idx = tid; idx < 256; idx += TPB) S.c1w[idx / 16][idx % 16] = c1_w[idx];
  if (tid == 0) S.c2b = c2_b[0];

  // ---- initial states h0/c0 broadcast over batch ----
  for (int idx = tid; idx < NLAY * BT * WW * 16; idx += TPB) {
    int c = idx & 15, w = (idx >> 4) % WW, b = (idx >> 4) / WW % BT, l = (idx >> 4) / (WW * BT);
    S.C[l][b][w][c]        = c0[(l * 16 + c) * WW + w];
    S.H[l][0][b][w + 1][c] = f2bf(h0[(l * 16 + c) * WW + w]);
  }
  __syncthreads();

  // conv B-fragment and bias are row-invariant: keep resident in VGPRs for the whole kernel
  Frag bwc; load_bfrag(bwc, &S.CWB[chn][0], 0, half);
  const float cbv = S.convb[chn];

  float lsum = 0.0f;

  for (int r = 0; r < HH; ++r) {
    // ---- stream x row r into rolling window (slot r&3), prefetch next row ----
    for (int idx = tid; idx < BT * 28; idx += TPB) {
      int b = idx / 28, c = idx % 28;
      const float* xp = &xg[((size_t)(b0 + b) * 29 + r) * 28 + c];
      S.XR[r & 3][b][3 + c] = f2bf(*xp);
      if (r + 1 < 29) __builtin_prefetch(xp + 28, 0, 0);   // -> global_prefetch_b8
    }
    __syncthreads();

    // ---- masked 7x7 conv row via WMMA: A[b][k=kh*8+kw]=x window, B=CWB, D=[b][16ch] ----
    for (int wi = 0; wi < 7; ++wi) {
      int w = wv * 7 + wi;
      Frag a;
#pragma unroll
      for (int g = 0; g < 8; ++g) {
        int k  = ((g >> 2) << 4) + (half << 3) + ((g & 3) << 1);
        int kh = k >> 3, kw = k & 7;
        int slot = (r - 3 + kh + 4) & 3;                 // rows r-3..r (invalid -> zeroed slot)
        const unsigned short* p = &S.XR[slot][chn][w + kw];
        unsigned lo = p[0], hi = p[1];                   // u16 pair (w-dependent alignment)
        a.u[g] = lo | (hi << 16);
      }
      v8f d;
#pragma unroll
      for (int i = 0; i < 8; ++i) d[i] = cbv;
      d = __builtin_amdgcn_wmma_f32_16x16x32_bf16(false, a.v, false, bwc.v, (short)0, d, false, false);
#pragma unroll
      for (int i = 0; i < 8; ++i)
        S.IN[0][i + (half << 3)][w + 1][chn] = f2bf(d[i]);
    }
    __syncthreads();

    // ---- 3 row-LSTM layers: gates = W[64x96] . [x_nbhd | h_nbhd], K=96 = 3 WMMA k-steps ----
    const int cur = r & 1, nxt = cur ^ 1;
#pragma unroll
    for (int l = 0; l < NLAY; ++l) {
      const int rb = l & 1, wb = rb ^ 1;                 // IN ping-pong: 0->1->0->1
      Frag bw[3][4];
#pragma unroll
      for (int ks = 0; ks < 3; ++ks)
#pragma unroll
        for (int nt = 0; nt < 4; ++nt)
          load_bfrag(bw[ks][nt], &S.Wl[l][nt * 16 + chn][0], ks, half);

      const int w0 = wv * 7;
      Frag aA[3], aB[3];
      {
        const unsigned short* xb = &S.IN[rb][chn][w0][0];   // 48 contiguous: [w-1..w+1]x16ch
        const unsigned short* hb = &S.H[l][cur][chn][w0][0];
#pragma unroll
        for (int ks = 0; ks < 3; ++ks) load_afrag_lstm(aA[ks], xb, hb, ks, half);
      }
      // software-pipelined over the 7 w-positions owned by this wave (even/odd double buffer)
      for (int p = 0; p < 3; ++p) {
        const int w = w0 + 2 * p;
        lstm_w_step(S, l, rb, wb, cur, nxt, chn, half, w,     w + 1, bw, aA, aB);
        lstm_w_step(S, l, rb, wb, cur, nxt, chn, half, w + 1, w + 2, bw, aB, aA);
      }
      lstm_w_step(S, l, rb, wb, cur, nxt, chn, half, w0 + 6, -1, bw, aA, aB);
      __syncthreads();
    }

    // ---- head: relu -> 1x1(16x16) relu -> 1x1(1x16) -> BCE-with-logits partial sum ----
    for (int idx = tid; idx < BT * WW; idx += TPB) {
      int b = idx / WW, w = idx % WW;
      float vin[16];
#pragma unroll
      for (int c = 0; c < 16; ++c) vin[c] = fmaxf(bf2f(S.IN[1][b][w + 1][c]), 0.0f);
      float z = S.c2b;
#pragma unroll
      for (int o = 0; o < 16; ++o) {
        float a1 = S.c1b[o];
#pragma unroll
        for (int c = 0; c < 16; ++c) a1 += S.c1w[o][c] * vin[c];
        z += S.c2w[o] * fmaxf(a1, 0.0f);
      }
      float tv = xg[((size_t)(b0 + b) * 29 + (r + 1)) * 28 + w];   // t = x[:,0,1:,:]
      lsum += fmaxf(z, 0.0f) - z * tv + __logf(1.0f + __expf(-fabsf(z)));
    }
    __syncthreads();
  }

  // ---- block reduction of the loss partial ----
  S.red[tid] = lsum;
  __syncthreads();
  if (tid == 0) {
    float s = 0.0f;
    for (int i = 0; i < TPB; ++i) s += S.red[i];
    partial[blockIdx.x] = s;
  }
}

__global__ void finalize_sum(const float* __restrict__ partial, float* __restrict__ out,
                             int nblk, float invn)
{
  __shared__ float sm[256];
  float s = 0.0f;
  for (int i = threadIdx.x; i < nblk; i += blockDim.x) s += partial[i];
  sm[threadIdx.x] = s;
  __syncthreads();
  if (threadIdx.x == 0) {
    float t = 0.0f;
    for (int i = 0; i < 256; ++i) t += sm[i];
    out[0] = t * invn;
  }
}

extern "C" void kernel_launch(void* const* d_in, const int* in_sizes, int n_in,
                              void* d_out, int out_size, void* d_ws, size_t ws_size,
                              hipStream_t stream)
{
  const float* xg     = (const float*)d_in[0];
  const float* conv_w = (const float*)d_in[1];
  const float* conv_b = (const float*)d_in[2];
  const float* is_w   = (const float*)d_in[3];
  const float* is_b   = (const float*)d_in[4];
  const float* ss_w   = (const float*)d_in[5];
  const float* ss_b   = (const float*)d_in[6];
  const float* h0     = (const float*)d_in[7];
  const float* c0     = (const float*)d_in[8];
  const float* c1_w   = (const float*)d_in[9];
  const float* c1_b   = (const float*)d_in[10];
  const float* c2_w   = (const float*)d_in[11];
  const float* c2_b   = (const float*)d_in[12];

  const int B    = in_sizes[0] / (29 * 28);
  const int nblk = B / BT;
  float* partial = (float*)d_ws;

  const size_t smem = sizeof(Smem);
  hipFuncSetAttribute(reinterpret_cast<const void*>(pixelrnn_fused),
                      hipFuncAttributeMaxDynamicSharedMemorySize, (int)smem);

  pixelrnn_fused<<<nblk, TPB, smem, stream>>>(xg, conv_w, conv_b, is_w, is_b, ss_w, ss_b,
                                              h0, c0, c1_w, c1_b, c2_w, c2_b, partial);

  const float invn = 1.0f / ((float)B * 28.0f * 28.0f);
  finalize_sum<<<1, 256, 0, stream>>>(partial, (float*)d_out, nblk, invn);
}
